// KPConvTransitionUp_46136538694258
// MI455X (gfx1250) — matrix operand order repaired and checked by reading, not compiled, not run on hardware
//
#include <hip/hip_runtime.h>
#include <hip/hip_bf16.h>

// ---------------- problem constants (from reference) ----------------
#define BATCH   8
#define N1      2048
#define N2      8192
#define CIN1    1024
#define CIN2    512
#define CO      512
#define NGROUPS 32
#define CPG     16          // CO / NGROUPS
#define KNN_K   3
#define EPS_KNN 1e-8f
#define GN_EPS  1e-5f

// ---------------- WMMA types ----------------
typedef __attribute__((ext_vector_type(16))) __bf16 v16bf;
typedef __attribute__((ext_vector_type(8)))  float  v8f;

union Frag {
    v16bf        v;
    unsigned int u[8];
};

__device__ __forceinline__ float relu_f(float x) { return x > 0.f ? x : 0.f; }

// packs two f32 into one dword of 2x bf16 -> v_cvt_pk_bf16_f32
__device__ __forceinline__ unsigned int pack_bf16(float a, float b)
{
    union { __bf16 h[2]; unsigned int u; } t;
    t.h[0] = (__bf16)a;
    t.h[1] = (__bf16)b;
    return t.u;
}

// ====================================================================
// 1) KNN top-3 + inverse-distance weights
//    grid: (N2/256, B), block 256
// ====================================================================
__global__ void knn_kernel(const float* __restrict__ p1,
                           const float* __restrict__ p2,
                           int* __restrict__ idx_out,
                           float* __restrict__ wgt_out)
{
    __shared__ float sp1[N1 * 3];
    const int b = blockIdx.y;
    const int j = blockIdx.x * blockDim.x + threadIdx.x;   // query point in [0,N2)

    const float* p1b = p1 + (size_t)b * N1 * 3;
    for (int i = threadIdx.x; i < N1 * 3; i += blockDim.x) sp1[i] = p1b[i];
    __syncthreads();

    const float qx = p2[((size_t)b * N2 + j) * 3 + 0];
    const float qy = p2[((size_t)b * N2 + j) * 3 + 1];
    const float qz = p2[((size_t)b * N2 + j) * 3 + 2];

    float d0 = 3.4e38f, d1 = 3.4e38f, d2 = 3.4e38f;
    int   i0 = 0, i1 = 0, i2 = 0;

    for (int i = 0; i < N1; ++i) {
        const float dx = qx - sp1[3 * i + 0];
        const float dy = qy - sp1[3 * i + 1];
        const float dz = qz - sp1[3 * i + 2];
        const float d  = dx * dx + dy * dy + dz * dz;
        if (d < d0)      { d2 = d1; i2 = i1; d1 = d0; i1 = i0; d0 = d; i0 = i; }
        else if (d < d1) { d2 = d1; i2 = i1; d1 = d;  i1 = i; }
        else if (d < d2) { d2 = d;  i2 = i; }
    }

    const float r0 = 1.f / (d0 + EPS_KNN);
    const float r1 = 1.f / (d1 + EPS_KNN);
    const float r2 = 1.f / (d2 + EPS_KNN);
    const float rs = 1.f / (r0 + r1 + r2);

    const size_t o = ((size_t)b * N2 + j) * 3;
    idx_out[o + 0] = i0; idx_out[o + 1] = i1; idx_out[o + 2] = i2;
    wgt_out[o + 0] = r0 * rs; wgt_out[o + 1] = r1 * rs; wgt_out[o + 2] = r2 * rs;
}

// ====================================================================
// 2) GEMM: out = W (CO x Kdim) * X[b] (Kdim x N), bf16 WMMA, f32 accum.
//    Block tile 128(M) x 128(N), 8 waves in a 4(M) x 2(N) grid;
//    each wave computes 32(M) x 64(N) = 2x4 WMMA subtiles per K-step.
//    LDS tiles hold bf16 PAIRS as dwords: row = 16 data uints + 2 pad.
//      As[m][p]  : pair p = K (2p, 2p+1) of row m        (A frag: v + kg*8)
//      Bs[n][p]  : pair p = K (2p, 2p+1) of column n     (B frag: v + kg*8)
//    transposed==1: out[b][n][CO] (f1t);  else out[b][CO][N] (raw2).
//    grid: (N/128, CO/128, B), block 256
// ====================================================================
#define BM  128
#define BNT 128
#define BK  32
#define AW  18              // uints per LDS row (16 data + 2 pad, 72B, 8B-aligned)

__global__ void gemm_bf16_kernel(const float* __restrict__ Wm,
                                 const float* __restrict__ X,
                                 float* __restrict__ out,
                                 int Kdim, int N, int transposed)
{
    __shared__ unsigned int As[BM][AW];
    __shared__ unsigned int Bs[BNT][AW];

    const int b     = blockIdx.z;
    const int n0    = blockIdx.x * BNT;
    const int m0    = blockIdx.y * BM;
    const int tid   = threadIdx.x;
    const int wave  = tid >> 5;
    const int lane  = tid & 31;
    const int lan16 = lane & 15;
    const int kg    = lane >> 4;          // lane half selector
    const int wm    = wave >> 1;          // 0..3  -> M offset wm*32
    const int wn    = wave & 1;           // 0..1  -> N offset wn*64

    const float* Xb = X + (size_t)b * Kdim * N;

    v8f acc[2][4];
    #pragma unroll
    for (int mi = 0; mi < 2; ++mi)
        #pragma unroll
        for (int s = 0; s < 4; ++s) acc[mi][s] = (v8f)(0.f);

    for (int k0 = 0; k0 < Kdim; k0 += BK) {
        // ---- stage A tile: 128 x 32, float4 loads, packed bf16 dword pairs ----
        #pragma unroll
        for (int i = 0; i < 4; ++i) {
            const int e = tid + i * 256;          // e in [0, 128*8)
            const int m = e >> 3, q = e & 7;      // q*4 = k offset
            const float4 f = *(const float4*)&Wm[(size_t)(m0 + m) * Kdim + (k0 + q * 4)];
            uint2 u;
            u.x = pack_bf16(f.x, f.y);
            u.y = pack_bf16(f.z, f.w);
            *(uint2*)&As[m][q * 2] = u;
        }
        // ---- stage B tile: 32(K) x 128(N), transpose through registers ----
        #pragma unroll
        for (int i = 0; i < 4; ++i) {
            const int e  = tid + i * 256;         // e in [0, 128*8)
            const int n  = e & 127, kq = e >> 7;  // kq*4 = k offset
            const float* px = &Xb[(size_t)(k0 + kq * 4) * N + (n0 + n)];
            const float f0 = px[0];
            const float f1 = px[(size_t)N];
            const float f2 = px[(size_t)2 * N];
            const float f3 = px[(size_t)3 * N];
            uint2 u;
            u.x = pack_bf16(f0, f1);
            u.y = pack_bf16(f2, f3);
            *(uint2*)&Bs[n][kq * 2] = u;
        }
        __syncthreads();

        // ---- A fragments (16x32 bf16 each): lane half kg -> K offset +16 ----
        Frag a[2];
        #pragma unroll
        for (int mi = 0; mi < 2; ++mi) {
            const int am = wm * 32 + mi * 16 + lan16;
            #pragma unroll
            for (int v = 0; v < 4; ++v) {
                a[mi].u[v]     = As[am][v + kg * 8];
                a[mi].u[4 + v] = As[am][4 + v + kg * 8];
            }
        }

        // ---- 4 B fragments, each reused by 2 M-subtiles: 8 WMMAs ----
        #pragma unroll
        for (int s = 0; s < 4; ++s) {
            Frag bf;
            const int bn = wn * 64 + s * 16 + lan16;
            #pragma unroll
            for (int v = 0; v < 8; ++v)
                bf.u[v] = Bs[bn][v + kg * 8];
            #pragma unroll
            for (int mi = 0; mi < 2; ++mi)
                acc[mi][s] = __builtin_amdgcn_wmma_f32_16x16x32_bf16(
                                 false, a[mi].v, false, bf.v, (short)0,
                                 acc[mi][s], false, false);
        }
        __syncthreads();
    }

    // ---- epilogue: C layout -> lanes 0-15: M=r, N=lane; 16-31: M=r+8 ----
    #pragma unroll
    for (int mi = 0; mi < 2; ++mi) {
        const int mrow = m0 + wm * 32 + mi * 16 + kg * 8;  // 8 consecutive M per lane
        #pragma unroll
        for (int s = 0; s < 4; ++s) {
            const int n = n0 + wn * 64 + s * 16 + lan16;
            if (transposed) {
                float* o = out + ((size_t)b * N + n) * CO + mrow;
                float4 lo, hi;
                lo.x = acc[mi][s][0]; lo.y = acc[mi][s][1];
                lo.z = acc[mi][s][2]; lo.w = acc[mi][s][3];
                hi.x = acc[mi][s][4]; hi.y = acc[mi][s][5];
                hi.z = acc[mi][s][6]; hi.w = acc[mi][s][7];
                *(float4*)(o)     = lo;
                *(float4*)(o + 4) = hi;
            } else {
                #pragma unroll
                for (int r = 0; r < 8; ++r)
                    out[((size_t)b * CO + (mrow + r)) * N + n] = acc[mi][s][r];
            }
        }
    }
}

// ====================================================================
// 3) GroupNorm statistics: one block per (b, group); 256 threads.
//    transposed==1: data[b][n][CO]; else data[b][CO][N].
//    stats[(b*32+g)*2] = mean, [+1] = rsqrt(var + eps)
// ====================================================================
__global__ void gn_stats_kernel(const float* __restrict__ data,
                                float* __restrict__ stats,
                                int N, int transposed)
{
    __shared__ float red1[256], red2[256];
    const int bg = blockIdx.x;
    const int b  = bg >> 5;
    const int g  = bg & 31;
    const int tid = threadIdx.x;

    float s1 = 0.f, s2 = 0.f;
    if (transposed) {
        for (int n = tid; n < N; n += 256) {
            const float* p = data + ((size_t)b * N + n) * CO + g * CPG;
            #pragma unroll
            for (int q = 0; q < 4; ++q) {
                const float4 f = *(const float4*)&p[q * 4];
                s1 += f.x + f.y + f.z + f.w;
                s2 += f.x * f.x + f.y * f.y + f.z * f.z + f.w * f.w;
            }
        }
    } else {
        const float* base = data + ((size_t)b * CO + g * CPG) * N;
        #pragma unroll 1
        for (int c = 0; c < CPG; ++c) {
            const float* row = base + (size_t)c * N;
            for (int n = tid; n < N; n += 256) { const float v = row[n]; s1 += v; s2 += v * v; }
        }
    }
    red1[tid] = s1; red2[tid] = s2;
    __syncthreads();
    for (int off = 128; off > 0; off >>= 1) {
        if (tid < off) { red1[tid] += red1[tid + off]; red2[tid] += red2[tid + off]; }
        __syncthreads();
    }
    if (tid == 0) {
        const float cnt = (float)CPG * (float)N;
        const float mu  = red1[0] / cnt;
        const float var = red2[0] / cnt - mu * mu;
        stats[(size_t)bg * 2 + 0] = mu;
        stats[(size_t)bg * 2 + 1] = rsqrtf(var + GN_EPS);
    }
}

// ====================================================================
// 4) Fused epilogue:
//    y[b,c,n2] = relu(gn2(raw2)) + sum_k w_k * relu(gn1(f1t[b, idx_k, c]))
//    grid: (N2/128, B), block 128
// ====================================================================
__global__ void fuse_out_kernel(const float* __restrict__ raw2,
                                const float* __restrict__ f1t,
                                const int* __restrict__ idx,
                                const float* __restrict__ wgt,
                                const float* __restrict__ st1,
                                const float* __restrict__ st2,
                                const float* __restrict__ gamma1,
                                const float* __restrict__ beta1,
                                const float* __restrict__ gamma2,
                                const float* __restrict__ beta2,
                                float* __restrict__ y)
{
    __shared__ float sg1[CO], sb1[CO], sg2[CO], sb2[CO];
    __shared__ float smu1[NGROUPS], srs1[NGROUPS], smu2[NGROUPS], srs2[NGROUPS];

    const int b  = blockIdx.y;
    const int n2 = blockIdx.x * blockDim.x + threadIdx.x;

    for (int i = threadIdx.x; i < CO; i += blockDim.x) {
        sg1[i] = gamma1[i]; sb1[i] = beta1[i];
        sg2[i] = gamma2[i]; sb2[i] = beta2[i];
    }
    for (int i = threadIdx.x; i < NGROUPS; i += blockDim.x) {
        smu1[i] = st1[((size_t)b * NGROUPS + i) * 2 + 0];
        srs1[i] = st1[((size_t)b * NGROUPS + i) * 2 + 1];
        smu2[i] = st2[((size_t)b * NGROUPS + i) * 2 + 0];
        srs2[i] = st2[((size_t)b * NGROUPS + i) * 2 + 1];
    }
    __syncthreads();

    const size_t qo = ((size_t)b * N2 + n2) * 3;
    const int   i0 = idx[qo + 0], i1 = idx[qo + 1], i2 = idx[qo + 2];
    const float w0 = wgt[qo + 0], w1 = wgt[qo + 1], w2 = wgt[qo + 2];

    const float* f1b = f1t + (size_t)b * N1 * CO;
    const float* c0  = f1b + (size_t)i0 * CO;
    const float* c1  = f1b + (size_t)i1 * CO;
    const float* c2  = f1b + (size_t)i2 * CO;
    const float* r2b = raw2 + (size_t)b * CO * N2 + n2;
    float*       yb  = y    + (size_t)b * CO * N2 + n2;

    #pragma unroll 4
    for (int c = 0; c < CO; ++c) {
        const int g = c >> 4;
        const float v   = r2b[(size_t)c * N2];
        float acc = relu_f((v - smu2[g]) * srs2[g] * sg2[c] + sb2[c]);
        const float m1 = smu1[g], r1 = srs1[g], gg = sg1[c], bb = sb1[c];
        acc += w0 * relu_f((c0[c] - m1) * r1 * gg + bb);
        acc += w1 * relu_f((c1[c] - m1) * r1 * gg + bb);
        acc += w2 * relu_f((c2[c] - m1) * r1 * gg + bb);
        yb[(size_t)c * N2] = acc;
    }
}

// ====================================================================
// host side
// ====================================================================
extern "C" void kernel_launch(void* const* d_in, const int* in_sizes, int n_in,
                              void* d_out, int out_size, void* d_ws, size_t ws_size,
                              hipStream_t stream)
{
    const float* p1 = (const float*)d_in[0];
    const float* x1 = (const float*)d_in[1];
    const float* p2 = (const float*)d_in[2];
    const float* x2 = (const float*)d_in[3];
    const float* W1 = (const float*)d_in[4];
    const float* g1 = (const float*)d_in[5];
    const float* b1 = (const float*)d_in[6];
    const float* W2 = (const float*)d_in[7];
    const float* g2 = (const float*)d_in[8];
    const float* b2 = (const float*)d_in[9];

    // workspace layout (bytes)
    char* ws = (char*)d_ws;
    const size_t sz_f1t  = (size_t)BATCH * N1 * CO * sizeof(float);      //  32 MB
    const size_t sz_raw2 = (size_t)BATCH * CO * N2 * sizeof(float);      // 128 MB
    const size_t sz_idx  = (size_t)BATCH * N2 * KNN_K * sizeof(int);
    const size_t sz_wgt  = (size_t)BATCH * N2 * KNN_K * sizeof(float);
    const size_t sz_st   = (size_t)BATCH * NGROUPS * 2 * sizeof(float);

    float* f1t   = (float*)(ws);
    float* raw2  = (float*)(ws + sz_f1t);
    int*   idx   = (int*)  (ws + sz_f1t + sz_raw2);
    float* wgt   = (float*)(ws + sz_f1t + sz_raw2 + sz_idx);
    float* st1   = (float*)(ws + sz_f1t + sz_raw2 + sz_idx + sz_wgt);
    float* st2   = (float*)(ws + sz_f1t + sz_raw2 + sz_idx + sz_wgt + sz_st);

    float* out_p2 = (float*)d_out;                       // B*N2*3 floats
    float* out_y  = out_p2 + (size_t)BATCH * N2 * 3;     // B*CO*N2 floats

    // output 0: p2 pass-through
    hipMemcpyAsync(out_p2, p2, (size_t)BATCH * N2 * 3 * sizeof(float),
                   hipMemcpyDeviceToDevice, stream);

    // 1) KNN
    {
        dim3 grid(N2 / 256, BATCH);
        knn_kernel<<<grid, 256, 0, stream>>>(p1, p2, idx, wgt);
    }
    // 2a) GEMM1: f1t[b][n][CO] = (W1 @ x1)^T
    {
        dim3 grid(N1 / BNT, CO / BM, BATCH);
        gemm_bf16_kernel<<<grid, 256, 0, stream>>>(W1, x1, f1t, CIN1, N1, 1);
    }
    // 2b) GEMM2: raw2[b][CO][N2] = W2 @ x2
    {
        dim3 grid(N2 / BNT, CO / BM, BATCH);
        gemm_bf16_kernel<<<grid, 256, 0, stream>>>(W2, x2, raw2, CIN2, N2, 0);
    }
    // 3) GroupNorm stats
    gn_stats_kernel<<<BATCH * NGROUPS, 256, 0, stream>>>(f1t,  st1, N1, 1);
    gn_stats_kernel<<<BATCH * NGROUPS, 256, 0, stream>>>(raw2, st2, N2, 0);
    // 4) fused epilogue -> y
    {
        dim3 grid(N2 / 128, BATCH);
        fuse_out_kernel<<<grid, 128, 0, stream>>>(raw2, f1t, idx, wgt,
                                                  st1, st2, g1, b1, g2, b2, out_y);
    }
}
